// Encoder_70995809403109
// MI455X (gfx1250) — compile-verified, matrix operand order checked by reading, not run on hardware
//
#include <hip/hip_runtime.h>
#include <hip/hip_bf16.h>
#include <math.h>

// ---------------------------------------------------------------------------
// GCN encoder for MI455X (gfx1250, wave32, WMMA).
//   per layer: hlin = H @ W   (WMMA f16->f32, LDS-staged fragments)
//              agg  = b + selfloop + sum_e norm_e * hlin[col_e]   (L2 atomics)
//              h    = tanh(alpha_c * agg + beta_c)                (GraphNorm folded)
// Assumes n_rows % 16 == 0 (N = 100000 = 6250 * 16).
// ---------------------------------------------------------------------------

typedef __attribute__((ext_vector_type(16))) _Float16 v16h;
typedef __attribute__((ext_vector_type(8)))  _Float16 v8h;
typedef __attribute__((ext_vector_type(8)))  float    v8f;

#define EPSV 1e-5f

// ---------------- degree / dinv ----------------

__global__ void k_deg_init(float* __restrict__ deg, int n) {
    for (int i = blockIdx.x * blockDim.x + threadIdx.x; i < n; i += gridDim.x * blockDim.x)
        deg[i] = 1.0f;  // self-loop contributes 1 to every node's degree
}

__global__ void k_deg_accum(const int* __restrict__ rowIdx, float* __restrict__ deg, int nE) {
    for (int e = blockIdx.x * blockDim.x + threadIdx.x; e < nE; e += gridDim.x * blockDim.x)
        atomicAdd(&deg[rowIdx[e]], 1.0f);
}

__global__ void k_dinv(float* __restrict__ deg, int n) {
    for (int i = blockIdx.x * blockDim.x + threadIdx.x; i < n; i += gridDim.x * blockDim.x) {
        float d = deg[i];
        deg[i] = (d > 0.0f) ? rsqrtf(d) : 0.0f;
    }
}

// ---------------- WMMA GEMM: Y[n x FO] = H[n x FI] @ W[FI x FO] ----------------
// One block = 16-row stripe; FO/16 waves, one 16x16 tile each.
// A-tile and W^T staged in LDS as f16, K zero-padded to a multiple of 32.
// Fragment layout (ISA 16-bit A, wave32): packed pairs sit at consecutive K,
// so each 16-element fragment is two contiguous 8xf16 (b128) LDS loads.

template <int FI, int FO>
__global__ __launch_bounds__(32 * (FO / 16)) void k_gemm_wmma(const float* __restrict__ H,
                                                              const float* __restrict__ W,
                                                              float* __restrict__ Y) {
    constexpr int KP = (FI + 31) & ~31;   // K padded: 8->32, 64->64, 128->128
    constexpr int NW = FO / 16;           // waves per block
    constexpr int NT = 32 * NW;           // threads per block

    __shared__ alignas(16) _Float16 sA[16 * KP];       // A tile, row-major [r][k]
    __shared__ alignas(16) _Float16 sWt[FO * KP];      // W transposed  [n][k]

    const int tid = threadIdx.x;
    const long rowBase = (long)blockIdx.x * 16;

    // stage W^T (zero-pad k >= FI)
    for (int i = tid; i < FO * KP; i += NT) {
        int n = i / KP, k = i % KP;
        sWt[i] = (k < FI) ? (_Float16)W[k * FO + n] : (_Float16)0.0f;
    }
    // stage A tile (zero-pad k >= FI)
    for (int i = tid; i < 16 * KP; i += NT) {
        int r = i / KP, k = i % KP;
        sA[i] = (k < FI) ? (_Float16)H[(rowBase + r) * FI + k] : (_Float16)0.0f;
    }
    __syncthreads();

    const int wv   = tid >> 5;
    const int lane = tid & 31;
    const int half = lane >> 4;
    const int l16  = lane & 15;

    v8f acc = {};
#pragma unroll
    for (int k0 = 0; k0 < KP; k0 += 32) {
        const _Float16* pa = &sA[l16 * KP + k0 + half * 8];
        const _Float16* pb = &sWt[(wv * 16 + l16) * KP + k0 + half * 8];
        v8h alo = *(const v8h*)pa;          // fragment elems 0..7  : K = k0+8h .. +7
        v8h ahi = *(const v8h*)(pa + 16);   // fragment elems 8..15 : K = k0+16+8h .. +7
        v8h blo = *(const v8h*)pb;
        v8h bhi = *(const v8h*)(pb + 16);
        v16h a, b;
#pragma unroll
        for (int i = 0; i < 8; ++i) {
            a[i] = alo[i];  a[8 + i] = ahi[i];
            b[i] = blo[i];  b[8 + i] = bhi[i];
        }
        acc = __builtin_amdgcn_wmma_f32_16x16x32_f16(
            /*neg_a=*/false, a, /*neg_b=*/false, b,
            /*c_mod=*/(short)0, acc, /*reuse_a=*/false, /*reuse_b=*/false);
    }

    // C/D layout: VGPR v -> row = 16*tm + v + 8*half, col = 16*wv + l16
    float* yp = Y + (rowBase + 8 * half) * FO + wv * 16 + l16;
#pragma unroll
    for (int v = 0; v < 8; ++v)
        yp[(long)v * FO] = acc[v];
}

// ---------------- aggregation ----------------
// agg[i,c] = bias[c] + dinv[i]^2 * hlin[i,c]   (self-loop + bias, full overwrite)

template <int FO>
__global__ void k_agg_init(const float* __restrict__ Hlin, const float* __restrict__ dinv,
                           const float* __restrict__ bias, float* __restrict__ Agg, int n_rows) {
    long total = (long)n_rows * FO;
    for (long idx = (long)blockIdx.x * blockDim.x + threadIdx.x; idx < total;
         idx += (long)gridDim.x * blockDim.x) {
        int i = (int)(idx / FO);
        int c = (int)(idx % FO);
        float di = dinv[i];
        Agg[idx] = bias[c] + di * di * Hlin[idx];
    }
}

// One wave per edge; lane covers FO/32 consecutive channels (vector gather).
// hlin (<=51 MB) is L2-resident on MI455X (192 MB L2) -> gathers + atomics hit L2.
template <int FO>
__global__ __launch_bounds__(256) void k_spmm(const float* __restrict__ Hlin,
                                              const int* __restrict__ rowIdx,
                                              const int* __restrict__ colIdx,
                                              const float* __restrict__ dinv,
                                              float* __restrict__ Agg, int nEdges) {
    constexpr int VEC = FO / 32;
    const int lane = threadIdx.x & 31;
    int wave = (blockIdx.x * blockDim.x + threadIdx.x) >> 5;
    int nWaves = (gridDim.x * blockDim.x) >> 5;
    for (int e = wave; e < nEdges; e += nWaves) {
        int r = rowIdx[e];
        int c = colIdx[e];
        float nm = dinv[r] * dinv[c];
        const float* src = Hlin + (long)c * FO + lane * VEC;
        float* dst = Agg + (long)r * FO + lane * VEC;
        if constexpr (VEC == 4) {
            float4 v = *(const float4*)src;           // global_load_b128
            atomicAdd(dst + 0, nm * v.x);
            atomicAdd(dst + 1, nm * v.y);
            atomicAdd(dst + 2, nm * v.z);
            atomicAdd(dst + 3, nm * v.w);
        } else {
            float2 v = *(const float2*)src;           // global_load_b64
            atomicAdd(dst + 0, nm * v.x);
            atomicAdd(dst + 1, nm * v.y);
        }
    }
}

// ---------------- GraphNorm stats ----------------

__global__ void k_zero(float* __restrict__ p, int n) {
    int i = blockIdx.x * blockDim.x + threadIdx.x;
    if (i < n) p[i] = 0.0f;
}

// blockDim == FO; thread c owns channel c; grid-stride over rows; per-block atomics.
template <int FO>
__global__ void k_colsum(const float* __restrict__ Agg, float* __restrict__ stats, int n_rows) {
    int c = threadIdx.x;
    float s = 0.0f, s2 = 0.0f;
    for (int i = blockIdx.x; i < n_rows; i += gridDim.x) {
        float v = Agg[(long)i * FO + c];
        s += v;
        s2 += v * v;
    }
    atomicAdd(&stats[c], s);
    atomicAdd(&stats[FO + c], s2);
}

// Fold GraphNorm into per-channel affine: out = alpha*h + beta
// var = E[h^2] - 2*gm*mean^2 + gm^2*mean^2
template <int FO>
__global__ void k_stats_fin(const float* __restrict__ stats, const float* __restrict__ gw,
                            const float* __restrict__ gb, const float* __restrict__ gm,
                            float* __restrict__ ab, int n_rows) {
    int c = threadIdx.x;
    if (c >= FO) return;
    float inv_n = 1.0f / (float)n_rows;
    float mean = stats[c] * inv_n;
    float ex2  = stats[FO + c] * inv_n;
    float g    = gm[c];
    float var  = ex2 - 2.0f * g * mean * mean + g * g * mean * mean;
    float alpha = gw[c] * rsqrtf(var + EPSV);
    ab[c]      = alpha;
    ab[FO + c] = gb[c] - alpha * g * mean;
}

template <int FO>
__global__ void k_norm_tanh(const float* __restrict__ Agg, const float* __restrict__ ab,
                            float* __restrict__ Out, int n_rows) {
    long total = (long)n_rows * FO;
    for (long idx = (long)blockIdx.x * blockDim.x + threadIdx.x; idx < total;
         idx += (long)gridDim.x * blockDim.x) {
        int c = (int)(idx % FO);
        Out[idx] = tanhf(ab[c] * Agg[idx] + ab[FO + c]);
    }
}

// ---------------- host-side layer driver ----------------

template <int FI, int FO>
static void run_layer(const float* Hin, const float* W, const float* b,
                      const float* gw, const float* gb, const float* gm,
                      const int* rowIdx, const int* colIdx, const float* dinv,
                      float* hlin, float* agg, float* stats, float* ab,
                      float* Hout, int n, int nE, hipStream_t s) {
    const int stripes = n / 16;                 // n % 16 == 0 (N = 100000)
    k_gemm_wmma<FI, FO><<<stripes, 32 * (FO / 16), 0, s>>>(Hin, W, hlin);

    k_agg_init<FO><<<4096, 256, 0, s>>>(hlin, dinv, b, agg, n);
    k_spmm<FO><<<4096, 256, 0, s>>>(hlin, rowIdx, colIdx, dinv, agg, nE);

    k_zero<<<1, 2 * FO, 0, s>>>(stats, 2 * FO);
    k_colsum<FO><<<1024, FO, 0, s>>>(agg, stats, n);
    k_stats_fin<FO><<<1, FO, 0, s>>>(stats, gw, gb, gm, ab, n);
    k_norm_tanh<FO><<<4096, 256, 0, s>>>(agg, ab, Hout, n);
}

extern "C" void kernel_launch(void* const* d_in, const int* in_sizes, int n_in,
                              void* d_out, int out_size, void* d_ws, size_t ws_size,
                              hipStream_t stream) {
    const float* x    = (const float*)d_in[0];
    const int*   eidx = (const int*)d_in[1];
    const int N = in_sizes[2];          // batch has one entry per node
    const int E = in_sizes[1] / 2;      // edge_index is (2, E)
    const int* rowIdx = eidx;           // destinations
    const int* colIdx = eidx + E;       // sources

    // per-layer params: W,b,gw,gb,gm starting at index 3
    const float* P[4][5];
    for (int l = 0; l < 4; ++l)
        for (int j = 0; j < 5; ++j)
            P[l][j] = (const float*)d_in[3 + 5 * l + j];

    // workspace layout (256B aligned)
    size_t off = 0;
    auto take = [&](size_t bytes) -> char* {
        char* p = (char*)d_ws + off;
        off = (off + bytes + 255) & ~(size_t)255;
        return p;
    };
    float* dinv  = (float*)take((size_t)N * sizeof(float));
    float* bufA  = (float*)take((size_t)N * 128 * sizeof(float));
    float* bufB  = (float*)take((size_t)N * 128 * sizeof(float));
    float* stats = (float*)take(256 * sizeof(float));
    float* ab    = (float*)take(256 * sizeof(float));
    (void)ws_size;

    // symmetric normalization: deg (with self-loops) -> dinv, in place
    k_deg_init<<<(N + 255) / 256, 256, 0, stream>>>(dinv, N);
    k_deg_accum<<<2048, 256, 0, stream>>>(rowIdx, dinv, E);
    k_dinv<<<(N + 255) / 256, 256, 0, stream>>>(dinv, N);

    float* out = (float*)d_out;
    // Layer 0: x(N,8) -> bufB(N,64)  (hlin=bufA, agg=bufB, tanh in place)
    run_layer<8, 64>(x, P[0][0], P[0][1], P[0][2], P[0][3], P[0][4],
                     rowIdx, colIdx, dinv, bufA, bufB, stats, ab, bufB, N, E, stream);
    // Layer 1: bufB(N,64) -> bufB(N,128)
    run_layer<64, 128>(bufB, P[1][0], P[1][1], P[1][2], P[1][3], P[1][4],
                       rowIdx, colIdx, dinv, bufA, bufB, stats, ab, bufB, N, E, stream);
    // Layer 2: bufB(N,128) -> bufB(N,128)
    run_layer<128, 128>(bufB, P[2][0], P[2][1], P[2][2], P[2][3], P[2][4],
                        rowIdx, colIdx, dinv, bufA, bufB, stats, ab, bufB, N, E, stream);
    // Layer 3: bufB(N,128) -> d_out(N,128)
    run_layer<128, 128>(bufB, P[3][0], P[3][1], P[3][2], P[3][3], P[3][4],
                        rowIdx, colIdx, dinv, bufA, bufB, stats, ab, out, N, E, stream);
    (void)out_size; (void)n_in;
}
